// posatt_26439818674933
// MI455X (gfx1250) — matrix operand order repaired and checked by reading, not compiled, not run on hardware
//
#include <hip/hip_runtime.h>
#include <hip/hip_bf16.h>

#define Bsz 4
#define Lsz 2048
#define Dsz 64
#define NH  4
#define OUTW (Dsz + NH * Dsz)   // 320
#define KC  64                  // K-chunk staged in LDS per iteration
#define PI_F 3.14159265358979f

typedef __attribute__((ext_vector_type(16))) __bf16 v16bf;
typedef __attribute__((ext_vector_type(2)))  __bf16 v2bf;
typedef __attribute__((ext_vector_type(8)))  float  v8f;

// ---------------------------------------------------------------------------
// Phase 1: per-row (b,i) distance row, 0.2-quantile threshold t (interpolated
// order stats 409/410 of 2048), per-head softmax denominators Z, and copy of
// the inputs slice into out[:, :, 0:64].
// ---------------------------------------------------------------------------
__global__ __launch_bounds__(256)
void posatt_phase1(const float* __restrict__ mesh,
                   const float* __restrict__ inputs,
                   const float* __restrict__ lmda,
                   float* __restrict__ out,
                   float* __restrict__ wst,   // [B*L] thresholds
                   float* __restrict__ wsz)   // [B*H*L] denominators
{
    const int row = blockIdx.x;            // 0 .. B*L-1
    const int b   = row >> 11;             // /2048
    const int i   = row & (Lsz - 1);
    const int tid = threadIdx.x;

    const float2* meshv = (const float2*)mesh;
    const float2 pi2 = meshv[b * Lsz + i];

    // 8 distances per thread, strided for coalesced float2 loads
    float d[8];
#pragma unroll
    for (int k = 0; k < 8; ++k) {
        const int j = tid + k * 256;
        const float2 pj = meshv[b * Lsz + j];
        const float dx = pi2.x - pj.x, dy = pi2.y - pj.y;
        d[k] = dx * dx + dy * dy;
    }

    // Binary search on non-negative float bit patterns (monotone) for the
    // 410th and 411th smallest values (0-indexed order stats 409, 410).
    __shared__ int cnt;
    unsigned sel[2];
    const unsigned ranks[2] = {410u, 411u};
    for (int s = 0; s < 2; ++s) {
        unsigned lo = 0u, hi = 0x40800000u;   // 4.0f > max possible d = 2.0
        while (lo < hi) {
            const unsigned mid = (lo + hi) >> 1;
            if (tid == 0) cnt = 0;
            __syncthreads();
            int c = 0;
#pragma unroll
            for (int k = 0; k < 8; ++k) c += (__float_as_uint(d[k]) <= mid) ? 1 : 0;
            atomicAdd(&cnt, c);
            __syncthreads();
            const unsigned tot = (unsigned)cnt;
            __syncthreads();
            if (tot >= ranks[s]) hi = mid; else lo = mid + 1;
        }
        sel[s] = lo;
    }
    const float v0 = __uint_as_float(sel[0]);
    const float v1 = __uint_as_float(sel[1]);
    const float t  = v0 + 0.4f * (v1 - v0);   // pos = 0.2*(2048-1) = 409.4

    // Per-head scales (head-independent mask: scale_h > 0 commutes with quantile)
    float sc[NH];
#pragma unroll
    for (int h = 0; h < NH; ++h)
        sc[h] = tanf(0.25f * PI_F * (1.0f - 1e-7f) * (1.0f + sinf(lmda[h])));

    __shared__ float Zs[NH];
    if (tid < NH) Zs[tid] = 0.0f;
    __syncthreads();

    float zp[NH] = {0.f, 0.f, 0.f, 0.f};
#pragma unroll
    for (int k = 0; k < 8; ++k) {
        if (d[k] <= t) {
#pragma unroll
            for (int h = 0; h < NH; ++h) zp[h] += __expf(-sc[h] * d[k]);
        }
    }
#pragma unroll
    for (int h = 0; h < NH; ++h) atomicAdd(&Zs[h], zp[h]);  // ds_add_f32
    __syncthreads();

    if (tid == 0) wst[row] = t;
    if (tid < NH) wsz[(b * NH + tid) * Lsz + i] = Zs[tid];
    if (tid < Dsz)
        out[(size_t)row * OUTW + tid] = inputs[(size_t)row * Dsz + tid];
}

// ---------------------------------------------------------------------------
// Phase 2: conv[b,i,h,:] = sum_j att[b,h,i,j] * inputs[b,j,:] via bf16 WMMA.
// Each wave owns a 16-row x 64-col output strip: one A fragment per K-step,
// four independent WMMAs into four v8f accumulators. 8 waves -> 128 rows per
// block. K staged in LDS in KC=64 chunks, bf16-packed pairwise along K so the
// staging path is cvt_pk_bf16 + dword LDS stores.
// ---------------------------------------------------------------------------
__global__ __launch_bounds__(256)
void posatt_phase2(const float* __restrict__ mesh,
                   const float* __restrict__ inputs,
                   const float* __restrict__ lmda,
                   const float* __restrict__ wst,
                   const float* __restrict__ wsz,
                   float* __restrict__ out)
{
    const int b = blockIdx.z;
    const int h = blockIdx.y;
    const int rowbase = blockIdx.x * 128;
    const int tid  = threadIdx.x;
    const int lane = tid & 31;
    const int wave = tid >> 5;       // 0..7 -> row tile

    __shared__ __bf16 Bt[Dsz * KC];   // [col][k], k fastest; 8KB
    __shared__ float  meshch[2 * KC]; // 64 mesh points (x,y)

    const float scale  = tanf(0.25f * PI_F * (1.0f - 1e-7f) * (1.0f + sinf(lmda[h])));
    const float nscale = -scale;

    // Per-lane row constants (A-layout: M = lane & 15 for both half-waves)
    const int mrow = rowbase + wave * 16 + (lane & 15);
    const float2 pme = ((const float2*)mesh)[b * Lsz + mrow];
    const float t_i  = wst[b * Lsz + mrow];
    const float invZ = 1.0f / wsz[(b * NH + h) * Lsz + mrow];

    const int kbase = (lane < 16) ? 0 : 16;   // A: K start per half-wave
    const int koffB = (lane < 16) ? 0 : 16;   // B: K start per half-wave
    const int nlane = lane & 15;

    v8f acc0 = {}, acc1 = {}, acc2 = {}, acc3 = {};

    for (int kc = 0; kc < Lsz; kc += KC) {
        __syncthreads();
        // Stage inputs[b, kc:kc+KC, 0:64] -> Bt[col][k] as bf16, packed (k,k+1)
        // pairs so each store is one dword (v_cvt_pk_bf16_f32 + ds_store_b32).
#pragma unroll
        for (int q = 0; q < 8; ++q) {
            const int lin = q * 256 + tid;    // 0..2047
            const int c   = lin & 63;         // column (coalesced across lanes)
            const int kp  = lin >> 6;         // k-pair 0..31
            const float g0 = inputs[((size_t)(b * Lsz + kc + 2 * kp)) * Dsz + c];
            const float g1 = inputs[((size_t)(b * Lsz + kc + 2 * kp + 1)) * Dsz + c];
            v2bf pk;
            pk[0] = (__bf16)g0;
            pk[1] = (__bf16)g1;
            *(v2bf*)&Bt[c * KC + 2 * kp] = pk;
        }
        if (tid < 2 * KC) meshch[tid] = mesh[(size_t)(b * Lsz + kc) * 2 + tid];
        __syncthreads();

        if (kc + KC < Lsz)   // pull next inputs chunk toward L2 (global_prefetch_b8)
            __builtin_prefetch(&inputs[((size_t)(b * Lsz + kc + KC + (tid >> 2))) * Dsz], 0, 1);

#pragma unroll
        for (int ks = 0; ks < 2; ++ks) {
            // A fragment: att weights, built directly in contiguous-K order, same as B
            // (lane M = lane&15; elems 0..15 -> K kbase+0..15)
            v16bf a;
#pragma unroll
            for (int e = 0; e < 16; ++e) {
                const int kk = ks * 32 + kbase + e;
                const float dx = pme.x - meshch[2 * kk];
                const float dy = pme.y - meshch[2 * kk + 1];
                const float dist = dx * dx + dy * dy;
                const float w = (dist <= t_i) ? __expf(nscale * dist) * invZ : 0.0f;
                a[e] = (__bf16)w;
            }

            // Four B fragments (N tiles), contiguous K in LDS, 32B aligned
            const __bf16* bbase = &Bt[ks * 32 + koffB];
            const v16bf b0 = *(const v16bf*)(bbase + (0 * 16 + nlane) * KC);
            const v16bf b1 = *(const v16bf*)(bbase + (1 * 16 + nlane) * KC);
            const v16bf b2 = *(const v16bf*)(bbase + (2 * 16 + nlane) * KC);
            const v16bf b3 = *(const v16bf*)(bbase + (3 * 16 + nlane) * KC);

            acc0 = __builtin_amdgcn_wmma_f32_16x16x32_bf16(false, a, false, b0, (short)0, acc0, false, false);
            acc1 = __builtin_amdgcn_wmma_f32_16x16x32_bf16(false, a, false, b1, (short)0, acc1, false, false);
            acc2 = __builtin_amdgcn_wmma_f32_16x16x32_bf16(false, a, false, b2, (short)0, acc2, false, false);
            acc3 = __builtin_amdgcn_wmma_f32_16x16x32_bf16(false, a, false, b3, (short)0, acc3, false, false);
        }
    }

    // C layout: VGPR r -> M = mOff + r; N = lane & 15
    const int mOff = (lane < 16) ? 0 : 8;
#pragma unroll
    for (int r = 0; r < 8; ++r) {
        const int rrow = rowbase + wave * 16 + mOff + r;
        float* orow = &out[(size_t)(b * Lsz + rrow) * OUTW + Dsz + h * Dsz];
        orow[0 * 16 + nlane] = acc0[r];
        orow[1 * 16 + nlane] = acc1[r];
        orow[2 * 16 + nlane] = acc2[r];
        orow[3 * 16 + nlane] = acc3[r];
    }
}

// ---------------------------------------------------------------------------
extern "C" void kernel_launch(void* const* d_in, const int* in_sizes, int n_in,
                              void* d_out, int out_size, void* d_ws, size_t ws_size,
                              hipStream_t stream) {
    const float* mesh   = (const float*)d_in[0];   // (B, L, 2)
    const float* inputs = (const float*)d_in[1];   // (B, L, D)
    const float* lmda   = (const float*)d_in[2];   // (H, 1, 1)
    float* out = (float*)d_out;                    // (B, L, 320)

    float* wst = (float*)d_ws;                     // B*L thresholds   (32 KB)
    float* wsz = wst + (size_t)Bsz * Lsz;          // B*H*L denoms     (128 KB)

    posatt_phase1<<<dim3(Bsz * Lsz), dim3(256), 0, stream>>>(
        mesh, inputs, lmda, out, wst, wsz);

    posatt_phase2<<<dim3(Lsz / 128, NH, Bsz), dim3(256), 0, stream>>>(
        mesh, inputs, lmda, wst, wsz, out);
}
